// MultiHeadAttention_6201932775710
// MI455X (gfx1250) — matrix operand order repaired and checked by reading, not compiled, not run on hardware
//
#include <hip/hip_runtime.h>

// ---------------------------------------------------------------------------
// MHA forward for MI455X (gfx1250): f16 WMMA (v_wmma_f32_16x16x32_f16) with
// f32 accumulate, wave32.  Weights pre-packed to f16 col-major so every WMMA
// B-operand is one contiguous 32B load; weight panels staged into LDS with the
// Tensor Data Mover (tensor_load_to_lds + s_wait_tensorcnt).
// ---------------------------------------------------------------------------

typedef __attribute__((ext_vector_type(16))) _Float16 v16h;
typedef __attribute__((ext_vector_type(8)))  _Float16 v8h;
typedef __attribute__((ext_vector_type(8)))  float    v8f;
typedef __attribute__((ext_vector_type(4)))  unsigned int v4u;
typedef __attribute__((ext_vector_type(8)))  int      v8i;
typedef __attribute__((ext_vector_type(4)))  int      v4i;

#define D_MODEL 1024
#define N_HEADS 16
#define D_K     64
#define SEQ     2048
#define BATCH   2
#define BS      (BATCH * SEQ)       // 4096 tokens
#define BH      (BATCH * N_HEADS)   // 32 (b,h) pairs

static __device__ __forceinline__ v8f wmma_f16(v16h a, v16h b, v8f c) {
  return __builtin_amdgcn_wmma_f32_16x16x32_f16(false, a, false, b, (short)0, c,
                                                false, false);
}

// ---------------------------------------------------------------------------
// TDM: DMA a contiguous 64-row x 1024-elem f16 panel (128KB) into LDS offset 0.
// Descriptor per CDNA5 ISA ch.8: group0 {count=1, lds_addr, global_addr,
// type=2}; group1 {data_size=1(2B), tensor dims huge (no OOB), tile 1024x64,
// dim0 stride 1024}.
// ---------------------------------------------------------------------------
static __device__ __forceinline__ void tdm_load_panel_128k(const _Float16* gsrc) {
  unsigned long long ga = (unsigned long long)(const void*)gsrc;
  v4u g0;
  g0[0] = 1u;                                            // count=1, user mode
  g0[1] = 0u;                                            // lds_addr = 0 (dynamic LDS base)
  g0[2] = (unsigned)(ga & 0xFFFFFFFFu);                  // global_addr[31:0]
  g0[3] = (unsigned)((ga >> 32) & 0x01FFFFFFu) | (2u << 30);  // addr[56:32] | type=2
  v8i g1;
  const unsigned td = 1u << 20;                          // giant tensor dims -> no clip
  g1[0] = (int)(1u << 16);                               // data_size=1 (2 bytes)
  g1[1] = (int)((td & 0xFFFFu) << 16);                   // tensor_dim0[15:0]
  g1[2] = (int)((td >> 16) | ((td & 0xFFFFu) << 16));    // td0[31:16] | td1[15:0]
  g1[3] = (int)((td >> 16) | (1024u << 16));             // td1[31:16] | tile_dim0=1024
  g1[4] = 64;                                            // tile_dim1=64, tile_dim2=0
  g1[5] = 1024;                                          // tensor_dim0_stride=1024
  g1[6] = 0;
  g1[7] = 0;
  v4i z4 = {0, 0, 0, 0};
#if __has_builtin(__builtin_amdgcn_tensor_load_to_lds)
#if __clang_major__ >= 23
  v8i z8 = {0, 0, 0, 0, 0, 0, 0, 0};
  __builtin_amdgcn_tensor_load_to_lds(g0, g1, z4, z4, z8, 0);
#else
  __builtin_amdgcn_tensor_load_to_lds(g0, g1, z4, z4, 0);
#endif
  __builtin_amdgcn_s_wait_tensorcnt(0);
#else
  (void)g0; (void)g1; (void)z4;
#endif
}

// Stage a weight panel (64 cols of Wt, full K=1024) into LDS.
static __device__ __forceinline__ void stage_panel(const _Float16* panel,
                                                   _Float16* lds, int tid) {
#if __has_builtin(__builtin_amdgcn_tensor_load_to_lds)
  if (tid == 0) tdm_load_panel_128k(panel);
#else
  const v8h* src = (const v8h*)panel;   // panel is contiguous 65536 halfs
  v8h* dst = (v8h*)lds;
  for (int i = tid; i < 64 * 1024 / 8; i += 128) dst[i] = src[i];
#endif
  __syncthreads();
}

// ---------------------------------------------------------------------------
// Prep kernel A: convert Q/K/V activations f32 -> f16 (coalesced, 4/thread).
// grid (BS*D/1024, 1, 3), block 256.
// ---------------------------------------------------------------------------
__global__ void cvt_x_kernel(const float* __restrict__ Q,
                             const float* __restrict__ Kin,
                             const float* __restrict__ V,
                             _Float16* __restrict__ xq,
                             _Float16* __restrict__ xk,
                             _Float16* __restrict__ xv) {
  const int which = blockIdx.z;
  const float* src = (which == 0) ? Q : (which == 1) ? Kin : V;
  _Float16* dst = (which == 0) ? xq : (which == 1) ? xk : xv;
  const size_t i = ((size_t)blockIdx.x * blockDim.x + threadIdx.x) * 4;
  float4 f = *(const float4*)(src + i);
  dst[i + 0] = (_Float16)f.x;
  dst[i + 1] = (_Float16)f.y;
  dst[i + 2] = (_Float16)f.z;
  dst[i + 3] = (_Float16)f.w;
}

// ---------------------------------------------------------------------------
// Prep kernel B: pack weights W[k][j] f32 -> Wt[j][k] f16 (LDS transpose).
// grid (32, 32, 4), block (32, 8).
// ---------------------------------------------------------------------------
__global__ void w_pack_kernel(const float* __restrict__ wq, const float* __restrict__ wk,
                              const float* __restrict__ wv, const float* __restrict__ wo,
                              _Float16* __restrict__ wtq, _Float16* __restrict__ wtk,
                              _Float16* __restrict__ wtv, _Float16* __restrict__ wto) {
  __shared__ float t[32][33];
  const int which = blockIdx.z;
  const float* W = (which == 0) ? wq : (which == 1) ? wk : (which == 2) ? wv : wo;
  _Float16* Wt   = (which == 0) ? wtq : (which == 1) ? wtk : (which == 2) ? wtv : wto;
  const int bx = blockIdx.x * 32;   // k-tile base (row of W)
  const int by = blockIdx.y * 32;   // j-tile base (col of W)
  for (int i = threadIdx.y; i < 32; i += 8)
    t[i][threadIdx.x] = W[(size_t)(bx + i) * D_MODEL + by + threadIdx.x];
  __syncthreads();
  for (int i = threadIdx.y; i < 32; i += 8)
    Wt[(size_t)(by + i) * D_MODEL + bx + threadIdx.x] = (_Float16)t[threadIdx.x][i];
}

// ---------------------------------------------------------------------------
// Shared f16 GEMM main loop: acc(16x16) += A(16x32) x B(32x16) over K=1024.
// A: per-lane contiguous 16B global loads.  B: 32B LDS loads from TDM panel.
// ---------------------------------------------------------------------------
static __device__ __forceinline__ v8f gemm_mainloop(const _Float16* __restrict__ arow,
                                                    const _Float16* lds,
                                                    int wave, int g, int r) {
  v8f acc = {};
  const _Float16* brow = lds + (size_t)(wave * 16 + r) * D_MODEL;  // lane's Wt row
  for (int k0 = 0; k0 < D_MODEL; k0 += 32) {
    v8h lo = *(const v8h*)(arow + k0 + 16 * g);         // K = k0 + 16g + e
    v8h hi = *(const v8h*)(arow + k0 + 16 * g + 8);     // K = k0 + 16g + 8 + e
    v16h A;
#pragma unroll
    for (int e = 0; e < 8; ++e) { A[e] = lo[e]; A[8 + e] = hi[e]; }
    v16h Bv = *(const v16h*)(brow + k0 + 16 * g);       // K = k0 + 16g + e
    acc = wmma_f16(A, Bv, acc);
  }
  return acc;
}

// ---------------------------------------------------------------------------
// Kernel 1: fused Q/K/V projection.  grid (D/64, BS/16, 3), block 128,
// dynamic LDS 128KB (weight panel via TDM).  Q/K stored split-heads f16;
// V stored transposed vt[bh][d][s] for contiguous P@V B-tiles.
// ---------------------------------------------------------------------------
__global__ void qkv_proj_kernel(const _Float16* __restrict__ xq,
                                const _Float16* __restrict__ xk,
                                const _Float16* __restrict__ xv,
                                const _Float16* __restrict__ wtq,
                                const _Float16* __restrict__ wtk,
                                const _Float16* __restrict__ wtv,
                                const float* __restrict__ bq,
                                const float* __restrict__ bk,
                                const float* __restrict__ bv,
                                _Float16* __restrict__ qh,
                                _Float16* __restrict__ kh,
                                _Float16* __restrict__ vt) {
  extern __shared__ _Float16 lds[];   // 64 x 1024 f16 weight panel
  const int tid = threadIdx.x;
  const int wave = tid >> 5;
  const int lane = tid & 31;
  const int g = lane >> 4;
  const int r = lane & 15;

  const int which = blockIdx.z;
  const _Float16* X  = (which == 0) ? xq  : (which == 1) ? xk  : xv;
  const _Float16* Wt = (which == 0) ? wtq : (which == 1) ? wtk : wtv;
  const float* bias  = (which == 0) ? bq  : (which == 1) ? bk  : bv;

  const int row0   = blockIdx.y * 16;
  const int colblk = blockIdx.x * 64;
  const int col0   = colblk + wave * 16;

  stage_panel(Wt + (size_t)colblk * D_MODEL, lds, tid);

  v8f acc = gemm_mainloop(X + (size_t)(row0 + r) * D_MODEL, lds, wave, g, r);

  const int jcol = col0 + r;
  const float bval = bias[jcol];
  const int h = jcol >> 6;
  const int d = jcol & (D_K - 1);
  if (which == 2) {
    // vt[((b*H + h)*dk + d) * SEQ + s]
#pragma unroll
    for (int i = 0; i < 8; ++i) {
      const int tok = row0 + i + 8 * g;
      const int b = tok >> 11;
      const int s = tok & (SEQ - 1);
      vt[((size_t)(b * N_HEADS + h) * D_K + d) * SEQ + s] = (_Float16)(acc[i] + bval);
    }
  } else {
    _Float16* outh = (which == 0) ? qh : kh;
#pragma unroll
    for (int i = 0; i < 8; ++i) {
      const int tok = row0 + i + 8 * g;
      const int b = tok >> 11;
      const int s = tok & (SEQ - 1);
      outh[((size_t)(b * N_HEADS + h) * SEQ + s) * D_K + d] = (_Float16)(acc[i] + bval);
    }
  }
}

// ---------------------------------------------------------------------------
// Kernel 2: attention for one (b,h, 16-query tile).  grid (SEQ/16, BH),
// block 128.  LDS: full 16x2048 f32 score strip (128KB) + reductions.
// ---------------------------------------------------------------------------
__global__ void attn_kernel(const _Float16* __restrict__ qh,
                            const _Float16* __restrict__ kh,
                            const _Float16* __restrict__ vt,
                            float* __restrict__ attnW,
                            _Float16* __restrict__ aoh) {
  extern __shared__ char smem[];
  float* sc     = (float*)smem;                           // [16][SEQ]
  float* red    = (float*)(smem + (size_t)16 * SEQ * 4);  // [16][8]
  float* rowmax = red + 16 * 8;
  float* rinv   = rowmax + 16;

  const int tid  = threadIdx.x;
  const int wave = tid >> 5;
  const int lane = tid & 31;
  const int g = lane >> 4;
  const int r = lane & 15;

  const int bh = blockIdx.y;
  const int q0 = blockIdx.x * 16;

  // ---- Phase 1: scores = q k^T / 8 ----
  const _Float16* qrow = qh + ((size_t)bh * SEQ + q0 + r) * D_K;
  v16h Aq[2];
#pragma unroll
  for (int ks = 0; ks < 2; ++ks) {
    v8h lo = *(const v8h*)(qrow + ks * 32 + 16 * g);
    v8h hi = *(const v8h*)(qrow + ks * 32 + 16 * g + 8);
#pragma unroll
    for (int e = 0; e < 8; ++e) { Aq[ks][e] = lo[e]; Aq[ks][8 + e] = hi[e]; }
  }

  const int nbase = wave * (SEQ / 4);
  for (int t = 0; t < SEQ / 4; t += 16) {
    const int n0 = nbase + t;
    const _Float16* krow = kh + ((size_t)bh * SEQ + n0 + r) * D_K + 16 * g;
    v8f s = {};
#pragma unroll
    for (int ks = 0; ks < 2; ++ks) {
      v16h Bv = *(const v16h*)(krow + ks * 32);
      s = wmma_f16(Aq[ks], Bv, s);
    }
#pragma unroll
    for (int i = 0; i < 8; ++i)
      sc[(size_t)(i + 8 * g) * SEQ + n0 + r] = s[i] * 0.125f;
  }
  __syncthreads();

  // ---- Phase 2: row max ----
  const int srow = tid >> 3;
  const int seg  = tid & 7;
  {
    const float* p = sc + (size_t)srow * SEQ + seg * (SEQ / 8);
    float m = -3.402823e38f;
    for (int c = 0; c < SEQ / 8; ++c) m = fmaxf(m, p[c]);
    red[srow * 8 + seg] = m;
  }
  __syncthreads();
  if (tid < 16) {
    float m = red[tid * 8];
#pragma unroll
    for (int i = 1; i < 8; ++i) m = fmaxf(m, red[tid * 8 + i]);
    rowmax[tid] = m;
  }
  __syncthreads();

  // ---- Phase 3: exp in place + row sum ----
  {
    float* p = sc + (size_t)srow * SEQ + seg * (SEQ / 8);
    const float m = rowmax[srow];
    float sum = 0.f;
    for (int c = 0; c < SEQ / 8; ++c) { float e = __expf(p[c] - m); p[c] = e; sum += e; }
    red[srow * 8 + seg] = sum;
  }
  __syncthreads();
  if (tid < 16) {
    float s = 0.f;
#pragma unroll
    for (int i = 0; i < 8; ++i) s += red[tid * 8 + i];
    rinv[tid] = 1.0f / s;
  }
  __syncthreads();

  // ---- Phase 5: stream attn_weights (coalesced, written once) ----
  {
    float* dst = attnW + ((size_t)bh * SEQ + q0) * SEQ;
    for (int l = tid; l < 16 * SEQ; l += 128)
      dst[l] = sc[l] * rinv[l >> 11];
  }

  // ---- Phase 6: attn_out = P @ V  (V transposed -> contiguous B tiles) ----
  const int d0 = wave * 16;
  const float rsc = rinv[r];
  const _Float16* vrow = vt + ((size_t)bh * D_K + d0 + r) * SEQ;  // lane's d-row
  v8f acc = {};
  for (int kk = 0; kk < SEQ; kk += 32) {
    const float* ps = sc + (size_t)r * SEQ + kk + 16 * g;
    v16h A;
#pragma unroll
    for (int e = 0; e < 8; ++e) {
      A[e]     = (_Float16)(ps[e]      * rsc);
      A[8 + e] = (_Float16)(ps[8 + e] * rsc);
    }
    v16h Bv = *(const v16h*)(vrow + kk + 16 * g);   // key = kk + 16g + e
    acc = wmma_f16(A, Bv, acc);
  }

  const int bb = bh >> 4;
  const int hh = bh & (N_HEADS - 1);
#pragma unroll
  for (int i = 0; i < 8; ++i) {
    const size_t tok = (size_t)bb * SEQ + q0 + i + 8 * g;
    aoh[tok * D_MODEL + hh * D_K + d0 + r] = (_Float16)acc[i];
  }
}

// ---------------------------------------------------------------------------
// Kernel 3: output projection.  grid (D/64, BS/16), block 128, LDS 128KB.
// ---------------------------------------------------------------------------
__global__ void out_proj_kernel(const _Float16* __restrict__ aoh,
                                const _Float16* __restrict__ wto,
                                const float* __restrict__ bo,
                                float* __restrict__ out) {
  extern __shared__ _Float16 lds[];
  const int tid = threadIdx.x;
  const int wave = tid >> 5;
  const int lane = tid & 31;
  const int g = lane >> 4;
  const int r = lane & 15;
  const int row0   = blockIdx.y * 16;
  const int colblk = blockIdx.x * 64;
  const int col0   = colblk + wave * 16;

  stage_panel(wto + (size_t)colblk * D_MODEL, lds, tid);

  v8f acc = gemm_mainloop(aoh + (size_t)(row0 + r) * D_MODEL, lds, wave, g, r);

  const float bval = bo[col0 + r];
#pragma unroll
  for (int i = 0; i < 8; ++i)
    out[(size_t)(row0 + i + 8 * g) * D_MODEL + col0 + r] = acc[i] + bval;
}

// ---------------------------------------------------------------------------
extern "C" void kernel_launch(void* const* d_in, const int* in_sizes, int n_in,
                              void* d_out, int out_size, void* d_ws, size_t ws_size,
                              hipStream_t stream) {
  const float* Q  = (const float*)d_in[0];
  const float* K  = (const float*)d_in[1];
  const float* V  = (const float*)d_in[2];
  const float* wq = (const float*)d_in[3];
  const float* bq = (const float*)d_in[4];
  const float* wk = (const float*)d_in[5];
  const float* bk = (const float*)d_in[6];
  const float* wv = (const float*)d_in[7];
  const float* bv = (const float*)d_in[8];
  const float* wo = (const float*)d_in[9];
  const float* bo = (const float*)d_in[10];

  float* out   = (float*)d_out;
  float* attnW = out + (size_t)BS * D_MODEL;

  // Workspace layout (halfs).  aoh aliases xq (dead after kernel 1).
  const size_t NX = (size_t)BS * D_MODEL;   // 4M halfs per activation buffer
  const size_t NW = (size_t)D_MODEL * D_MODEL;
  _Float16* xq  = (_Float16*)d_ws;
  _Float16* xk  = xq + NX;
  _Float16* xv  = xk + NX;
  _Float16* wtq = xv + NX;
  _Float16* wtk = wtq + NW;
  _Float16* wtv = wtk + NW;
  _Float16* wto = wtv + NW;
  _Float16* qh  = wto + NW;
  _Float16* kh  = qh + NX;
  _Float16* vt  = kh + NX;
  _Float16* aoh = xq;                       // alias: xq unused after proj

  cvt_x_kernel<<<dim3((unsigned)(NX / 1024), 1, 3), dim3(256), 0, stream>>>(
      Q, K, V, xq, xk, xv);
  w_pack_kernel<<<dim3(32, 32, 4), dim3(32, 8), 0, stream>>>(
      wq, wk, wv, wo, wtq, wtk, wtv, wto);

  const size_t panel_lds = (size_t)64 * D_MODEL * sizeof(_Float16);  // 128KB
  qkv_proj_kernel<<<dim3(D_MODEL / 64, BS / 16, 3), dim3(128), panel_lds, stream>>>(
      xq, xk, xv, wtq, wtk, wtv, bq, bk, bv, qh, kh, vt);

  const size_t attn_lds = (size_t)16 * SEQ * 4 + 16 * 8 * 4 + 16 * 4 + 16 * 4;
  attn_kernel<<<dim3(SEQ / 16, BH, 1), dim3(128), attn_lds, stream>>>(
      qh, kh, vt, attnW, aoh);

  out_proj_kernel<<<dim3(D_MODEL / 64, BS / 16, 1), dim3(128), panel_lds, stream>>>(
      aoh, wto, bo, out);
}
